// SparseBahdanauAttention_81612968558784
// MI455X (gfx1250) — matrix operand order repaired and checked by reading, not compiled, not run on hardware
//
#include <hip/hip_runtime.h>
#include <hip/hip_bf16.h>

// Problem sizes (fixed by the reference)
static constexpr int B = 32;
static constexpr int S = 2048;
static constexpr int H = 1024;

typedef __attribute__((ext_vector_type(16))) __bf16 v16bf;
typedef __attribute__((ext_vector_type(8)))  __bf16 v8bf;
typedef __attribute__((ext_vector_type(8)))  float  v8f;

// ---------------------------------------------------------------------------
// Kernel 0: one-time f32 -> bf16 conversion of Ua_w (4MB -> 2MB, L2 resident).
// Removes per-block cvt work and halves B-side L2 traffic in the hot GEMM.
// ---------------------------------------------------------------------------
__global__ __launch_bounds__(256) void prep_bf16_kernel(
    const float* __restrict__ w, __bf16* __restrict__ wb) {
  const int i = (blockIdx.x * 256 + threadIdx.x) * 4;
  float4 v = *(const float4*)(w + i);
  wb[i]     = (__bf16)v.x;
  wb[i + 1] = (__bf16)v.y;
  wb[i + 2] = (__bf16)v.z;
  wb[i + 3] = (__bf16)v.w;
}

// ---------------------------------------------------------------------------
// Kernel 1: query[b,o] = decoder_hidden[b,:] . Wa_w[o,:] + Wa_b[o]
// ---------------------------------------------------------------------------
__global__ __launch_bounds__(256) void query_kernel(
    const float* __restrict__ dh, const float* __restrict__ Wa_w,
    const float* __restrict__ Wa_b, float* __restrict__ query) {
  const int b = blockIdx.x;
  __shared__ float h_lds[H];
  for (int i = threadIdx.x; i < H; i += 256) h_lds[i] = dh[b * H + i];
  __syncthreads();
  for (int o = threadIdx.x; o < H; o += 256) {
    const float* wrow = Wa_w + (size_t)o * H;
    float acc = Wa_b[o];
    for (int h = 0; h < H; h += 4) {
      float4 w = *(const float4*)(wrow + h);
      acc += w.x * h_lds[h] + w.y * h_lds[h + 1] +
             w.z * h_lds[h + 2] + w.w * h_lds[h + 3];
    }
    query[b * H + o] = acc;
  }
}

// ---------------------------------------------------------------------------
// Kernel 2 (hot): fused  energy[b,s] = Va_w . tanh(enc[b,s,:]@Ua_w^T + Ua_b
//                                                  + query[b,:]) + Va_b
// 16-row tile / block, 8 waves x 128 N-columns each, bf16 WMMA -> f32 acc.
// Double-buffered bf16 A tile in LDS; pre-converted bf16 B from L2.
// ---------------------------------------------------------------------------
static constexpr int ASTR = 56;  // LDS row stride in bf16: 112B = 28 banks,
                                 // 16B-aligned, conflict-free b128 reads

__global__ __launch_bounds__(256) void energy_kernel(
    const float* __restrict__ enc, const __bf16* __restrict__ ua_bf,
    const float* __restrict__ Ua_b, const float* __restrict__ query,
    const float* __restrict__ Va_w, const float* __restrict__ Va_b,
    float* __restrict__ energy) {
  __shared__ __align__(16) __bf16 a_lds[2][16][ASTR];
  __shared__ float e_lds[16];

  const int tid  = threadIdx.x;
  const int lane = tid & 31;
  const int wave = tid >> 5;
  const int b    = blockIdx.x >> 7;          // 128 row-tiles per batch
  const int s0   = (blockIdx.x & 127) << 4;  // first of 16 rows
  const int n_wave = wave << 7;              // this wave's 128 columns

  if (tid < 16) e_lds[tid] = 0.0f;

  // WMMA lane mapping (CDNA5 ISA, 16-bit A 16x32 / B Kx16):
  const int mrow = lane & 15;                // A row / B column / C column
  const int ka0  = (lane < 16) ? 0 : 8;      // A: two contiguous runs of 8 k's
  const int kb0  = (lane < 16) ? 0 : 8;      // B: two contiguous runs of 8 k's

  const v8f vzero = {0.f, 0.f, 0.f, 0.f, 0.f, 0.f, 0.f, 0.f};
  v8f acc[8];
#pragma unroll
  for (int i = 0; i < 8; ++i) acc[i] = vzero;

  const float* enc_b = enc + ((size_t)(b * S + s0)) * H;

  // Cooperative stage of one 16x32 tile as bf16 (512 elems / 256 threads)
  const int sm0 = tid >> 5, sk0 = tid & 31;        // element tid
  const int sm1 = (tid + 256) >> 5, sk1 = tid & 31;  // element tid+256

  // Prologue: stage K-chunk 0 into buffer 0
  a_lds[0][sm0][sk0] = (__bf16)enc_b[(size_t)sm0 * H + sk0];
  a_lds[0][sm1][sk1] = (__bf16)enc_b[(size_t)sm1 * H + sk1];
  __syncthreads();

  for (int kci = 0; kci < H / 32; ++kci) {
    const int kc  = kci << 5;
    const int cur = kci & 1;

    // Stage next K-chunk into the other buffer (overlaps with compute)
    if (kci + 1 < H / 32) {
      const int kn = kc + 32;
      a_lds[cur ^ 1][sm0][sk0] = (__bf16)enc_b[(size_t)sm0 * H + kn + sk0];
      a_lds[cur ^ 1][sm1][sk1] = (__bf16)enc_b[(size_t)sm1 * H + kn + sk1];
      if (kci + 2 < H / 32)  // prefetch chunk after next -> global_prefetch_b8
        __builtin_prefetch(&enc_b[(size_t)sm0 * H + kn + 32 + sk0], 0, 1);
    }

    // A fragment: two aligned 16B LDS loads per lane
    v8bf alo = *(const v8bf*)&a_lds[cur][mrow][ka0];
    v8bf ahi = *(const v8bf*)&a_lds[cur][mrow][ka0 + 16];
    v16bf afrag = __builtin_shufflevector(alo, ahi, 0, 1, 2, 3, 4, 5, 6, 7,
                                          8, 9, 10, 11, 12, 13, 14, 15);

    // 8 column-chunks of 16 per wave; B direct from L2-resident bf16 Ua_w
#pragma unroll
    for (int nc = 0; nc < 8; ++nc) {
      const int n = n_wave + (nc << 4) + mrow;
      const __bf16* brow = ua_bf + (size_t)n * H + kc + kb0;  // B[k][n]=Ua_w[n][k]
      v8bf blo = *(const v8bf*)brow;
      v8bf bhi = *(const v8bf*)(brow + 16);
      v16bf bfrag = __builtin_shufflevector(blo, bhi, 0, 1, 2, 3, 4, 5, 6, 7,
                                            8, 9, 10, 11, 12, 13, 14, 15);
      acc[nc] = __builtin_amdgcn_wmma_f32_16x16x32_bf16(
          false, afrag, false, bfrag, (short)0, acc[nc], false, false);
    }
    __syncthreads();  // next buffer fully staged; this buffer free to overwrite
  }

  // Epilogue: bias + tanh + Va weighting, reduce over N
  float e_part[8];
#pragma unroll
  for (int r = 0; r < 8; ++r) e_part[r] = 0.0f;
#pragma unroll
  for (int nc = 0; nc < 8; ++nc) {
    const int n = n_wave + (nc << 4) + mrow;
    const float bias = Ua_b[n] + query[b * H + n];
    const float vw   = Va_w[n];
#pragma unroll
    for (int r = 0; r < 8; ++r)
      e_part[r] += tanhf(acc[nc][r] + bias) * vw;
  }
  // Cross-lane reduce over the 16 lanes sharing a C row group
#pragma unroll
  for (int r = 0; r < 8; ++r) {
#pragma unroll
    for (int mask = 1; mask < 16; mask <<= 1)
      e_part[r] += __shfl_xor(e_part[r], mask, 32);
  }
  if (mrow == 0) {
    const int rbase = (lane < 16) ? 0 : 8;  // C: VGPR r -> row r (+8 hi half)
#pragma unroll
    for (int r = 0; r < 8; ++r) atomicAdd(&e_lds[rbase + r], e_part[r]);
  }
  __syncthreads();
  if (tid < 16) energy[b * S + s0 + tid] = e_lds[tid] + Va_b[0];
}

// ---------------------------------------------------------------------------
// Kernel 3: sparsemax over S per batch row (bitonic sort + scan in LDS)
// ---------------------------------------------------------------------------
__global__ __launch_bounds__(256) void sparsemax_kernel(
    const float* __restrict__ energy, float* __restrict__ attn) {
  const int b = blockIdx.x;
  const int tid = threadIdx.x;
  __shared__ float zs[S];          // sorted (descending)
  __shared__ float cs[S];          // cumulative sum of sorted
  __shared__ float chunk_sum[256];
  __shared__ int   kz_s;
  __shared__ float tau_s;

  for (int i = tid; i < S; i += 256) zs[i] = energy[b * S + i];
  if (tid == 0) kz_s = 0;
  __syncthreads();

  // Bitonic sort, descending
  for (int ksz = 2; ksz <= S; ksz <<= 1) {
    for (int j = ksz >> 1; j > 0; j >>= 1) {
      for (int i = tid; i < S; i += 256) {
        int ixj = i ^ j;
        if (ixj > i) {
          float a = zs[i], c = zs[ixj];
          bool desc = ((i & ksz) == 0);
          if (desc ? (a < c) : (a > c)) { zs[i] = c; zs[ixj] = a; }
        }
      }
      __syncthreads();
    }
  }

  // Inclusive scan: 8 contiguous elems per thread + chunk-offset scan
  const int base = tid * 8;
  float loc[8];
  {
    float run = 0.0f;
#pragma unroll
    for (int r = 0; r < 8; ++r) { run += zs[base + r]; loc[r] = run; }
    chunk_sum[tid] = run;
  }
  __syncthreads();
  if (tid == 0) {
    float a = 0.0f;
    for (int t = 0; t < 256; ++t) { float v = chunk_sum[t]; chunk_sum[t] = a; a += v; }
  }
  __syncthreads();
  {
    const float off = chunk_sum[tid];
#pragma unroll
    for (int r = 0; r < 8; ++r) cs[base + r] = loc[r] + off;
  }
  __syncthreads();

  // Support size k_z = #{k : 1 + k*z_sorted[k-1] > cumsum[k-1]}
  int cnt = 0;
#pragma unroll
  for (int r = 0; r < 8; ++r) {
    const int k = base + r + 1;
    if (1.0f + (float)k * zs[base + r] > cs[base + r]) cnt++;
  }
  atomicAdd(&kz_s, cnt);
  __syncthreads();
  if (tid == 0) {
    const int kz = kz_s > 0 ? kz_s : 1;
    tau_s = (cs[kz - 1] - 1.0f) / (float)kz;
  }
  __syncthreads();

  const float tau = tau_s;
  for (int i = tid; i < S; i += 256) {
    const float v = energy[b * S + i] - tau;
    attn[b * S + i] = v > 0.0f ? v : 0.0f;
  }
}

// ---------------------------------------------------------------------------
// Kernel 4: context[b,h] = sum_s attn[b,s] * enc[b,s,h]  (skip zero weights)
// ---------------------------------------------------------------------------
__global__ __launch_bounds__(256) void context_kernel(
    const float* __restrict__ enc, const float* __restrict__ attn,
    float* __restrict__ ctx) {
  const int b = blockIdx.x >> 2;                       // H/256 = 4 blocks per b
  const int h = ((blockIdx.x & 3) << 8) + threadIdx.x;
  const float* ep = enc + (size_t)(b * S) * H + h;
  const float* ap = attn + b * S;
  float acc = 0.0f;
  for (int s = 0; s < S; ++s) {
    const float w = ap[s];
    if (w != 0.0f) acc += w * ep[(size_t)s * H];
  }
  ctx[b * H + h] = acc;
}

// ---------------------------------------------------------------------------
extern "C" void kernel_launch(void* const* d_in, const int* in_sizes, int n_in,
                              void* d_out, int out_size, void* d_ws, size_t ws_size,
                              hipStream_t stream) {
  (void)in_sizes; (void)n_in; (void)out_size; (void)ws_size;
  const float* enc  = (const float*)d_in[0];
  const float* dh   = (const float*)d_in[1];
  const float* Wa_w = (const float*)d_in[2];
  const float* Wa_b = (const float*)d_in[3];
  const float* Ua_w = (const float*)d_in[4];
  const float* Ua_b = (const float*)d_in[5];
  const float* Va_w = (const float*)d_in[6];
  const float* Va_b = (const float*)d_in[7];

  float* out_ctx  = (float*)d_out;          // [B, H]   (context, flattened first)
  float* out_attn = out_ctx + B * H;        // [B, S]   (attention weights)
  float*  query  = (float*)d_ws;            // [B, H] scratch
  float*  energy = query + B * H;           // [B, S] scratch
  __bf16* ua_bf  = (__bf16*)(energy + B * S);  // [H, H] bf16 scratch (2MB)

  prep_bf16_kernel<<<(H * H) / (256 * 4), 256, 0, stream>>>(Ua_w, ua_bf);
  query_kernel<<<B, 256, 0, stream>>>(dh, Wa_w, Wa_b, query);
  energy_kernel<<<B * (S / 16), 256, 0, stream>>>(enc, ua_bf, Ua_b, query,
                                                  Va_w, Va_b, energy);
  sparsemax_kernel<<<B, 256, 0, stream>>>(energy, out_attn);
  context_kernel<<<B * (H / 256), 256, 0, stream>>>(enc, out_attn, out_ctx);
}
